// Generator_79834852098574
// MI455X (gfx1250) — matrix operand order repaired and checked by reading, not compiled, not run on hardware
//
#include <hip/hip_runtime.h>
#include <math.h>

// ============================================================================
// StyleGAN2 generator forward for MI455X (gfx1250, wave32, WMMA bf16).
//
// d_in layout assumption: jax.tree_util.tree_flatten of setup_inputs() dict
// (sorted dict keys at every level, lists in index order):
//   [0..12]  noises (4,8,8,16,16,32,32,64,64,128,128,256,256 squared)
//   [13]     const [1,512,4,4]
//   [14..18] conv1: act_b, mod_b, mod_w, noise_w, w
//   [19..78] convs[j=0..11]: base=19+5j: act_b, mod_b, mod_w, noise_w, w
//   [79..94] mapping[l=0..7]: 79+2l=b, 80+2l=w
//   [95..98] to_rgb1: bias, mod_b, mod_w, w
//   [99..122] to_rgbs[j=0..5]: base=99+4j: bias, mod_b, mod_w, w
//   [123]    z [8,512]
//
// modulated conv == demod[b,o] * conv(x * s[b,i], w*scale): weights are
// batch-shared -> implicit GEMM (pixels x Cout), K = Cin looped over taps,
// v_wmma_f32_16x16x32_bf16, ping-pong double-buffered loads, and the
// modulated input is written zero-padded (halo ring) so all fragment loads
// are unconditional (no exec-mask predication in the hot loop).
// ============================================================================

typedef __bf16 bf16_t;
typedef __bf16 v16bf __attribute__((ext_vector_type(16)));
typedef float  v8f   __attribute__((ext_vector_type(8)));

#define WPB 8     // waves per block (256 threads)
#define TPB 256

#define WMMA_BF16(Afrag, Bfrag, Cfrag) \
  __builtin_amdgcn_wmma_f32_16x16x32_bf16(false, (Afrag), false, (Bfrag), (short)0, (Cfrag), false, false)

union FragU { v16bf v; uint4 q[2]; };

__device__ __forceinline__ float flrelu(float v) {
  return (v > 0.0f ? v : 0.2f * v) * 1.4142135623730951f;
}

// ------------------------------- mapping net --------------------------------
__global__ void pixelnorm_kernel(const float* __restrict__ z, float* __restrict__ x) {
  __shared__ float red[512];
  const int b = blockIdx.x, d = threadIdx.x;
  const float v = z[b * 512 + d];
  red[d] = v * v;
  __syncthreads();
  for (int s = 256; s > 0; s >>= 1) {
    if (d < s) red[d] += red[d + s];
    __syncthreads();
  }
  const float mean = red[0] * (1.0f / 512.0f);
  x[b * 512 + d] = v * rsqrtf(mean + 1e-8f);
}

__global__ void eqlinear_kernel(const float* __restrict__ x, const float* __restrict__ w,
                                const float* __restrict__ bias, float* __restrict__ y) {
  const int idx = blockIdx.x * blockDim.x + threadIdx.x;
  if (idx >= 8 * 512) return;
  const int b = idx >> 9, o = idx & 511;
  const float scale = 0.01f * 0.044194173824159216f;  // lr_mul / sqrt(512)
  float acc = bias[o] * 0.01f;
  const float* xb = x + b * 512;
  const float* wo = w + (size_t)o * 512;
  for (int d = 0; d < 512; ++d) acc += xb[d] * wo[d] * scale;
  y[idx] = flrelu(acc);
}

// ----------------------------- per-layer styles -----------------------------
__global__ void style_kernel(const float* __restrict__ lat, const float* __restrict__ mw,
                             const float* __restrict__ mb, float* __restrict__ s, int Cin) {
  const int idx = blockIdx.x * blockDim.x + threadIdx.x;
  if (idx >= 8 * Cin) return;
  const int b = idx / Cin, i = idx % Cin;
  const float scale = 0.044194173824159216f;  // 1/sqrt(512)
  float acc = mb[i];
  const float* lb = lat + b * 512;
  const float* wi = mw + (size_t)i * 512;
  for (int d = 0; d < 512; ++d) acc += lb[d] * wi[d] * scale;
  s[idx] = acc;
}

__global__ void demod_kernel(const float* __restrict__ w, const float* __restrict__ s,
                             float* __restrict__ dm, int Cin, int Cout, float scale2) {
  const int idx = blockIdx.x * blockDim.x + threadIdx.x;
  if (idx >= 8 * Cout) return;
  const int b = idx / Cout, o = idx % Cout;
  const float* wo = w + (size_t)o * Cin * 9;
  const float* sb = s + b * Cin;
  float sum = 0.0f;
  for (int i = 0; i < Cin; ++i) {
    float w2 = 0.0f;
#pragma unroll
    for (int t = 0; t < 9; ++t) { const float wv = wo[i * 9 + t]; w2 += wv * wv; }
    const float sv = sb[i];
    sum += w2 * sv * sv;
  }
  dm[idx] = rsqrtf(sum * scale2 + 1e-8f);
}

// Pack w[Cout,Cin,3,3]*scale into per-lane WMMA B-fragment layout (bf16):
// wp[ ((tap*KC+kci)*NT+nt)*512 + lane*16 + e ], K = kci*32 + 8*(lane>>4) + (e&7) + 16*(e>>3),
// col o = nt*16 + (lane&15).
__global__ void wpack_kernel(const float* __restrict__ w, bf16_t* __restrict__ wp,
                             int Cin, int Cout, float scale) {
  const int idx = blockIdx.x * blockDim.x + threadIdx.x;
  const int total = 9 * Cin * Cout;
  if (idx >= total) return;
  const int e = idx & 15;
  const int lane = (idx >> 4) & 31;
  const int NT = Cout >> 4, KC = Cin >> 5;
  int rest = idx >> 9;
  const int nt = rest % NT; rest /= NT;
  const int kci = rest % KC;
  const int tap = rest / KC;
  const int k = kci * 32 + ((lane >> 4) << 3) + (e & 7) + ((e >> 3) << 4);
  const int o = nt * 16 + (lane & 15);
  wp[idx] = (bf16_t)(w[((size_t)o * Cin + k) * 9 + tap] * scale);
}

__global__ void const_init_kernel(const float* __restrict__ c, bf16_t* __restrict__ act) {
  const int idx = blockIdx.x * blockDim.x + threadIdx.x;
  if (idx >= 8 * 16 * 512) return;
  const int ch = idx & 511;
  const int p = idx >> 9;
  const int yx = p & 15;
  act[idx] = (bf16_t)c[ch * 16 + yx];
}

// xm (padded (H+2)x(W+2) NHWC) = act[b,y,x,i] * s[b,i]; zero halo ring.
__global__ void premod_pad_kernel(const bf16_t* __restrict__ act, const float* __restrict__ s,
                                  bf16_t* __restrict__ xm, int C, int H, int W, int total) {
  const int idx = blockIdx.x * blockDim.x + threadIdx.x;
  if (idx >= total) return;
  const int c = idx % C;
  const int pix = idx / C;
  const int xp = pix % (W + 2);
  const int rest = pix / (W + 2);
  const int yp = rest % (H + 2);
  const int b = rest / (H + 2);
  const int y = yp - 1, x = xp - 1;
  float v = 0.0f;
  if (y >= 0 && y < H && x >= 0 && x < W)
    v = (float)act[(((size_t)b * H + y) * W + x) * C + c] * s[b * C + c];
  xm[idx] = (bf16_t)v;
}

// ----------------------- 3x3 same-res modulated conv ------------------------
// Implicit GEMM, 2 M-tiles x 4 N-tiles per wave.  Flattened (tap, K-chunk)
// loop, unrolled by 2 with ping-pong fragment buffers; padded input makes all
// A-fragment loads unconditional.
__global__ __launch_bounds__(TPB)
void conv3x3_wmma_kernel(const bf16_t* __restrict__ xmod, const bf16_t* __restrict__ wpack,
                         const float* __restrict__ dm, const float* __restrict__ noise,
                         const float* __restrict__ actb, const float* __restrict__ nwp,
                         bf16_t* __restrict__ out, int B, int H, int W, int Cin, int Cout) {
  const int lane = threadIdx.x & 31;
  const int wave = blockIdx.x * WPB + (threadIdx.x >> 5);
  const int numN = Cout >> 4;
  const int ng4  = numN >> 2;
  const int HW   = H * W;
  const int Wp   = W + 2, Hp = H + 2;
  const int numM2 = (B * HW) >> 5;       // pairs of 16-pixel M tiles
  if (wave >= numM2 * ng4) return;
  const int mt = (wave / ng4) << 1;      // first M tile of the pair
  const int ng = wave % ng4;

  const int r   = lane & 15;
  const int p0  = mt * 16 + r;
  const int p1  = p0 + 16;
  const int b0i = p0 / HW, py0 = (p0 % HW) / W, px0 = p0 % W;
  const int b1i = p1 / HW, py1 = (p1 % HW) / W, px1 = p1 % W;
  const int kb  = (lane >> 4) << 3;
  const int KC  = Cin >> 5;
  const int TOT = 9 * KC;                // always even (KC in {4,8,16})
  const v16bf* wp = (const v16bf*)wpack;

  v8f c00, c01, c02, c03, c10, c11, c12, c13;
#pragma unroll
  for (int i = 0; i < 8; ++i) {
    c00[i] = 0.f; c01[i] = 0.f; c02[i] = 0.f; c03[i] = 0.f;
    c10[i] = 0.f; c11[i] = 0.f; c12[i] = 0.f; c13[i] = 0.f;
  }

  // incremental "next fragment to load" state (padded coords: +dy+1, +dx+1)
  int tapn = 0, kcin = 0;
  const bf16_t* ap0;
  const bf16_t* ap1;
  auto tapBase = [&](int tap) {
    const int dy = tap / 3, dx = tap % 3;   // 0..2 == padded offset
    ap0 = xmod + (((size_t)b0i * Hp + (py0 + dy)) * Wp + (px0 + dx)) * Cin + kb;
    ap1 = xmod + (((size_t)b1i * Hp + (py1 + dy)) * Wp + (px1 + dx)) * Cin + kb;
  };
  auto loadCur = [&](FragU& f0, FragU& f1) {
    f0.q[0] = *(const uint4*)ap0;
    f0.q[1] = *(const uint4*)(ap0 + 16);
    f1.q[0] = *(const uint4*)ap1;
    f1.q[1] = *(const uint4*)(ap1 + 16);
  };
  auto advance = [&]() {
    ++kcin;
    if (kcin == KC) {
      kcin = 0;
      ++tapn;
      if (tapn < 9) tapBase(tapn);
    } else {
      ap0 += 32;
      ap1 += 32;
    }
  };
  auto mm8 = [&](const FragU& f0, const FragU& f1,
                 const v16bf& w0, const v16bf& w1, const v16bf& w2, const v16bf& w3) {
    c00 = WMMA_BF16(f0.v, w0, c00);
    c01 = WMMA_BF16(f0.v, w1, c01);
    c02 = WMMA_BF16(f0.v, w2, c02);
    c03 = WMMA_BF16(f0.v, w3, c03);
    c10 = WMMA_BF16(f1.v, w0, c10);
    c11 = WMMA_BF16(f1.v, w1, c11);
    c12 = WMMA_BF16(f1.v, w2, c12);
    c13 = WMMA_BF16(f1.v, w3, c13);
  };

  size_t wb = (size_t)(ng << 2) * 32 + lane;
  const size_t wstep = (size_t)numN * 32;

  // stage A <- fragment 0
  FragU aA0, aA1, aB0, aB1;
  v16bf wA0, wA1, wA2, wA3, wB0, wB1, wB2, wB3;
  tapBase(0);
  loadCur(aA0, aA1);
  wA0 = wp[wb]; wA1 = wp[wb + 32]; wA2 = wp[wb + 64]; wA3 = wp[wb + 96];
  advance(); wb += wstep;

#pragma unroll 1
  for (int it = 0; it < TOT - 2; it += 2) {
    // stage B <- fragment it+1, compute fragment it (A)
    loadCur(aB0, aB1);
    wB0 = wp[wb]; wB1 = wp[wb + 32]; wB2 = wp[wb + 64]; wB3 = wp[wb + 96];
    advance(); wb += wstep;
    mm8(aA0, aA1, wA0, wA1, wA2, wA3);
    // stage A <- fragment it+2, compute fragment it+1 (B)
    loadCur(aA0, aA1);
    wA0 = wp[wb]; wA1 = wp[wb + 32]; wA2 = wp[wb + 64]; wA3 = wp[wb + 96];
    advance(); wb += wstep;
    mm8(aB0, aB1, wB0, wB1, wB2, wB3);
  }
  // tail: A holds fragment TOT-2; load TOT-1 into B
  loadCur(aB0, aB1);
  wB0 = wp[wb]; wB1 = wp[wb + 32]; wB2 = wp[wb + 64]; wB3 = wp[wb + 96];
  mm8(aA0, aA1, wA0, wA1, wA2, wA3);
  mm8(aB0, aB1, wB0, wB1, wB2, wB3);

  const float nw  = nwp[0];
  const int colr  = lane & 15;
  const int mbase = (lane >> 4) << 3;
  // batch index is tile-uniform (HW is a multiple of 16)
  const int bb0 = (mt * 16) / HW,       rr0 = mt * 16 - bb0 * HW;
  const int bb1 = ((mt + 1) * 16) / HW, rr1 = (mt + 1) * 16 - bb1 * HW;
  auto epi = [&](const v8f& a, int nt, int bb, int rrbase) {
    const int o = nt * 16 + colr;
    const float ab = actb[o];
    const float d = dm[bb * Cout + o];
#pragma unroll
    for (int v = 0; v < 8; ++v) {
      const int rr = rrbase + mbase + v;
      const float val = a[v] * d + nw * noise[rr] + ab;
      out[((size_t)bb * HW + rr) * Cout + o] = (bf16_t)flrelu(val);
    }
  };
  epi(c00, (ng << 2) + 0, bb0, rr0); epi(c01, (ng << 2) + 1, bb0, rr0);
  epi(c02, (ng << 2) + 2, bb0, rr0); epi(c03, (ng << 2) + 3, bb0, rr0);
  epi(c10, (ng << 2) + 0, bb1, rr1); epi(c11, (ng << 2) + 1, bb1, rr1);
  epi(c12, (ng << 2) + 2, bb1, rr1); epi(c13, (ng << 2) + 3, bb1, rr1);
}

// --------------------- transposed conv (stride-2 upsample) ------------------
// t[2u+a, 2v+b] += x[u,v,:] w[:,a,b]; parity-uniform 16-pixel tiles so the
// weight tap is tile-uniform; per-tap K loop ping-pong double-buffered.
// Input is the padded xmod: per-lane column clamped into the zero halo, so
// loads are unconditional (overshoot lanes read zeros / are discarded).
__global__ __launch_bounds__(TPB)
void tconv_wmma_kernel(const bf16_t* __restrict__ xmod, const bf16_t* __restrict__ wpack,
                       const float* __restrict__ dm, bf16_t* __restrict__ tout,
                       int B, int Hin, int Cin, int Cout) {
  const int Ht = 2 * Hin + 1;
  const int Hp = Hin + 2;
  const int XT = (Hin + 16) >> 4;      // ceil((Hin+1)/16) tiles per parity row
  const int numN = Cout >> 4, ng4 = numN >> 2;
  const int lane = threadIdx.x & 31;
  const int wave = blockIdx.x * WPB + (threadIdx.x >> 5);
  const int numM = B * Ht * 2 * XT;
  if (wave >= numM * ng4) return;
  const int mt = wave / ng4, ng = wave % ng4;

  int t = mt;
  const int xt = t % XT; t /= XT;
  const int xpar = t & 1; t >>= 1;
  const int Y = t % Ht;
  const int b = t / Ht;

  const int r = lane & 15;
  const int X = xpar + 2 * (xt * 16 + r);
  const int kb = (lane >> 4) << 3;
  const int KC = Cin >> 5;             // even, >= 4
  const v16bf* wp = (const v16bf*)wpack;
  const size_t wstep = (size_t)numN * 32;

  v8f c0, c1, c2, c3;
#pragma unroll
  for (int i = 0; i < 8; ++i) { c0[i] = 0.f; c1[i] = 0.f; c2[i] = 0.f; c3[i] = 0.f; }

  auto mm4 = [&](const FragU& f, const v16bf& w0, const v16bf& w1,
                 const v16bf& w2, const v16bf& w3) {
    c0 = WMMA_BF16(f.v, w0, c0);
    c1 = WMMA_BF16(f.v, w1, c1);
    c2 = WMMA_BF16(f.v, w2, c2);
    c3 = WMMA_BF16(f.v, w3, c3);
  };

  for (int a2 = (Y & 1); a2 <= 2; a2 += 2) {
    const int u = (Y - a2) >> 1;
    if (u < 0 || u >= Hin) continue;
    for (int tb = xpar; tb <= 2; tb += 2) {
      const int v2 = X - tb;           // always even; >= -2
      int vv = v2 >> 1;                // >= -1
      if (vv > Hin) vv = Hin;          // clamp into padded range (zero column)
      const bf16_t* ap = xmod + (((size_t)b * Hp + (u + 1)) * Hp + (vv + 1)) * Cin + kb;
      const int tap = a2 * 3 + tb;
      size_t wb = ((size_t)(tap * KC) * numN + (ng << 2)) * 32 + lane;

      auto loadAfr = [&](FragU& f, int kci) {
        const bf16_t* p = ap + kci * 32;
        f.q[0] = *(const uint4*)p;
        f.q[1] = *(const uint4*)(p + 16);
      };

      FragU aA, aB;
      v16bf wA0, wA1, wA2, wA3, wB0, wB1, wB2, wB3;
      loadAfr(aA, 0);
      wA0 = wp[wb]; wA1 = wp[wb + 32]; wA2 = wp[wb + 64]; wA3 = wp[wb + 96];
      wb += wstep;

#pragma unroll 1
      for (int kci = 0; kci < KC - 2; kci += 2) {
        loadAfr(aB, kci + 1);
        wB0 = wp[wb]; wB1 = wp[wb + 32]; wB2 = wp[wb + 64]; wB3 = wp[wb + 96];
        wb += wstep;
        mm4(aA, wA0, wA1, wA2, wA3);
        loadAfr(aA, kci + 2);
        wA0 = wp[wb]; wA1 = wp[wb + 32]; wA2 = wp[wb + 64]; wA3 = wp[wb + 96];
        wb += wstep;
        mm4(aB, wB0, wB1, wB2, wB3);
      }
      loadAfr(aB, KC - 1);
      wB0 = wp[wb]; wB1 = wp[wb + 32]; wB2 = wp[wb + 64]; wB3 = wp[wb + 96];
      mm4(aA, wA0, wA1, wA2, wA3);
      mm4(aB, wB0, wB1, wB2, wB3);
    }
  }

  const int colr  = lane & 15;
  const int mbase = (lane >> 4) << 3;
  auto epi = [&](const v8f& a, int nt) {
    const int o = nt * 16 + colr;
    const float d = dm[b * Cout + o];
#pragma unroll
    for (int v = 0; v < 8; ++v) {
      const int Xm = xpar + 2 * (xt * 16 + mbase + v);
      if (Xm < Ht)
        tout[(((size_t)b * Ht + Y) * Ht + Xm) * Cout + o] = (bf16_t)(a[v] * d);
    }
  };
  epi(c0, (ng << 2) + 0);
  epi(c1, (ng << 2) + 1);
  epi(c2, (ng << 2) + 2);
  epi(c3, (ng << 2) + 3);
}

// 4x4 blur of t (on (Ho+1)^2, zero outside) + noise + bias + flrelu -> NHWC bf16
__global__ void blur_kernel(const bf16_t* __restrict__ t, const float* __restrict__ noise,
                            const float* __restrict__ actb, const float* __restrict__ nwp,
                            bf16_t* __restrict__ out, int B, int Ho, int C, int total) {
  const int idx = blockIdx.x * blockDim.x + threadIdx.x;
  if (idx >= total) return;
  const int c = idx % C;
  const int pix = idx / C;
  const int x = pix % Ho;
  const int rest = pix / Ho;
  const int y = rest % Ho;
  const int b = rest / Ho;
  const int Ht = Ho + 1;
  const float kk[4] = {1.f, 3.f, 3.f, 1.f};
  float sum = 0.0f;
#pragma unroll
  for (int fy = 0; fy < 4; ++fy) {
    const int sy = y - 1 + fy;
    if (sy < 0 || sy >= Ht) continue;
#pragma unroll
    for (int fx = 0; fx < 4; ++fx) {
      const int sx = x - 1 + fx;
      if (sx < 0 || sx >= Ht) continue;
      sum += kk[fy] * kk[fx] * (float)t[(((size_t)b * Ht + sy) * Ht + sx) * C + c];
    }
  }
  const float val = sum * (1.0f / 16.0f) + nwp[0] * noise[y * Ho + x] + actb[c];
  out[idx] = (bf16_t)flrelu(val);
}

// to_rgb (1x1 modulated conv, no demod) + upsampled-skip add.  last -> NCHW f32
__global__ void torgb_kernel(const bf16_t* __restrict__ act, const float* __restrict__ s,
                             const float* __restrict__ wr, const float* __restrict__ bias,
                             const float* __restrict__ skipin, float* __restrict__ skipout,
                             float* __restrict__ dout, int B, int H, int Cin,
                             int hasSkip, int toNCHW, float scale) {
  const int idx = blockIdx.x * blockDim.x + threadIdx.x;
  if (idx >= B * H * H) return;
  const int b = idx / (H * H);
  const int rr = idx % (H * H);
  const int y = rr / H, x = rr % H;
  float a0 = bias[0], a1 = bias[1], a2 = bias[2];
  const bf16_t* ap = act + (size_t)idx * Cin;
  const float* sb = s + b * Cin;
  for (int i = 0; i < Cin; ++i) {
    const float xv = (float)ap[i] * sb[i] * scale;
    a0 += xv * wr[i];
    a1 += xv * wr[Cin + i];
    a2 += xv * wr[2 * Cin + i];
  }
  if (hasSkip) {
    const int Hold = H >> 1;
    const float kk[4] = {1.f, 3.f, 3.f, 1.f};
#pragma unroll
    for (int fy = 0; fy < 4; ++fy) {
      const int sy = y - 2 + fy;
      if (sy < 0 || (sy & 1)) continue;
      const int u = sy >> 1;
      if (u >= Hold) continue;
#pragma unroll
      for (int fx = 0; fx < 4; ++fx) {
        const int sx = x - 2 + fx;
        if (sx < 0 || (sx & 1)) continue;
        const int v = sx >> 1;
        if (v >= Hold) continue;
        const float wgt = kk[fy] * kk[fx] * (1.0f / 16.0f);
        const float* sp = skipin + (((size_t)b * Hold + u) * Hold + v) * 3;
        a0 += wgt * sp[0]; a1 += wgt * sp[1]; a2 += wgt * sp[2];
      }
    }
  }
  if (toNCHW) {
    const size_t base = (size_t)b * 3 * H * H + rr;
    dout[base] = a0;
    dout[base + (size_t)H * H] = a1;
    dout[base + (size_t)2 * H * H] = a2;
  } else {
    float* op = skipout + (size_t)idx * 3;
    op[0] = a0; op[1] = a1; op[2] = a2;
  }
}

// ============================== host orchestration ==========================
static inline int cdiv(int a, int b) { return (a + b - 1) / b; }

extern "C" void kernel_launch(void* const* d_in, const int* in_sizes, int n_in,
                              void* d_out, int out_size, void* d_ws, size_t ws_size,
                              hipStream_t stream) {
  (void)in_sizes; (void)n_in; (void)out_size; (void)ws_size;
  char* ws = (char*)d_ws;
  size_t off = 0;
  auto wsalloc = [&](size_t bytes) -> void* {
    void* p = ws + off;
    off += (bytes + 255) & ~(size_t)255;
    return p;
  };
  const size_t ACT_BYTES  = (size_t)8 * 256 * 256 * 128 * sizeof(bf16_t);
  const size_t XMOD_BYTES = (size_t)8 * 258 * 258 * 128 * sizeof(bf16_t);  // padded
  const size_t T_BYTES    = (size_t)8 * 257 * 257 * 128 * sizeof(bf16_t);
  const size_t SKIP_BYTES = (size_t)8 * 256 * 256 * 3 * sizeof(float);
  const size_t WPK_BYTES  = (size_t)9 * 512 * 512 * sizeof(bf16_t);

  bf16_t* actA = (bf16_t*)wsalloc(ACT_BYTES);
  bf16_t* actB = (bf16_t*)wsalloc(ACT_BYTES);
  bf16_t* xmod = (bf16_t*)wsalloc(XMOD_BYTES);
  bf16_t* tbuf = (bf16_t*)wsalloc(T_BYTES);
  float* skipA = (float*)wsalloc(SKIP_BYTES);
  float* skipB = (float*)wsalloc(SKIP_BYTES);
  bf16_t* wpk  = (bf16_t*)wsalloc(WPK_BYTES);
  float* wlat = (float*)wsalloc(8 * 512 * sizeof(float));
  float* mtmp = (float*)wsalloc(8 * 512 * sizeof(float));
  float* sbuf = (float*)wsalloc(8 * 512 * sizeof(float));
  float* dmod = (float*)wsalloc(8 * 512 * sizeof(float));

  // ---- mapping network: z -> wlat ----
  const float* z = (const float*)d_in[123];
  pixelnorm_kernel<<<8, 512, 0, stream>>>(z, wlat);
  for (int l = 0; l < 8; ++l) {
    const float* lb = (const float*)d_in[79 + 2 * l];
    const float* lw = (const float*)d_in[80 + 2 * l];
    float* src = (l % 2 == 0) ? wlat : mtmp;
    float* dst = (l % 2 == 0) ? mtmp : wlat;
    eqlinear_kernel<<<cdiv(8 * 512, 256), 256, 0, stream>>>(src, lw, lb, dst);
  }
  // 8 layers -> result back in wlat

  auto run_styled = [&](const bf16_t* inAct, bf16_t* outAct, int Hin, int Cin, int Cout,
                        bool up, int pbase, int nidx) {
    const float* actb = (const float*)d_in[pbase + 0];
    const float* modb = (const float*)d_in[pbase + 1];
    const float* modw = (const float*)d_in[pbase + 2];
    const float* nwp  = (const float*)d_in[pbase + 3];
    const float* w    = (const float*)d_in[pbase + 4];
    const float* noise = (const float*)d_in[nidx];
    const float scale = 1.0f / sqrtf((float)(Cin * 9));
    style_kernel<<<cdiv(8 * Cin, 256), 256, 0, stream>>>(wlat, modw, modb, sbuf, Cin);
    demod_kernel<<<cdiv(8 * Cout, 256), 256, 0, stream>>>(w, sbuf, dmod, Cin, Cout,
                                                          1.0f / (float)(Cin * 9));
    wpack_kernel<<<cdiv(9 * Cin * Cout, 256), 256, 0, stream>>>(w, wpk, Cin, Cout, scale);
    const int ptotal = 8 * (Hin + 2) * (Hin + 2) * Cin;
    premod_pad_kernel<<<cdiv(ptotal, 256), 256, 0, stream>>>(inAct, sbuf, xmod, Cin,
                                                             Hin, Hin, ptotal);
    if (!up) {
      const int waves = ((8 * Hin * Hin) >> 5) * ((Cout >> 4) >> 2);
      conv3x3_wmma_kernel<<<cdiv(waves, WPB), TPB, 0, stream>>>(
          xmod, wpk, dmod, noise, actb, nwp, outAct, 8, Hin, Hin, Cin, Cout);
    } else {
      const int Ht = 2 * Hin + 1, XT = (Hin + 16) >> 4;
      const int waves = 8 * Ht * 2 * XT * ((Cout >> 4) >> 2);
      tconv_wmma_kernel<<<cdiv(waves, WPB), TPB, 0, stream>>>(
          xmod, wpk, dmod, tbuf, 8, Hin, Cin, Cout);
      const int Ho = 2 * Hin, total = 8 * Ho * Ho * Cout;
      blur_kernel<<<cdiv(total, 256), 256, 0, stream>>>(tbuf, noise, actb, nwp, outAct,
                                                        8, Ho, Cout, total);
    }
  };

  auto run_torgb = [&](const bf16_t* inAct, int H, int Cin, int pbase,
                       const float* skipin, float* skipout, int last) {
    const float* bias = (const float*)d_in[pbase + 0];
    const float* modb = (const float*)d_in[pbase + 1];
    const float* modw = (const float*)d_in[pbase + 2];
    const float* wr   = (const float*)d_in[pbase + 3];
    style_kernel<<<cdiv(8 * Cin, 256), 256, 0, stream>>>(wlat, modw, modb, sbuf, Cin);
    const float scale = 1.0f / sqrtf((float)Cin);
    const int total = 8 * H * H;
    torgb_kernel<<<cdiv(total, 256), 256, 0, stream>>>(
        inAct, sbuf, wr, bias, skipin, skipout, (float*)d_out, 8, H, Cin,
        skipin ? 1 : 0, last, scale);
  };

  // ---- conv1 @4x4 + to_rgb1 ----
  const_init_kernel<<<cdiv(8 * 16 * 512, 256), 256, 0, stream>>>((const float*)d_in[13], actA);
  run_styled(actA, actB, 4, 512, 512, false, 14, 0);
  run_torgb(actB, 4, 512, 95, nullptr, skipA, 0);

  // ---- 6 resolution doublings ----
  const int CH[7] = {512, 512, 512, 512, 512, 256, 128};
  bf16_t* cur = actB; bf16_t* other = actA;
  float* skcur = skipA; float* skother = skipB;
  int H = 4;
  for (int j = 0; j < 6; ++j) {
    const int Cin = CH[j], Cout = CH[j + 1];
    run_styled(cur, other, H, Cin, Cout, true, 19 + 5 * (2 * j), 2 * j + 1);
    H *= 2;
    { bf16_t* tmp = cur; cur = other; other = tmp; }
    run_styled(cur, other, H, Cout, Cout, false, 19 + 5 * (2 * j + 1), 2 * j + 2);
    { bf16_t* tmp = cur; cur = other; other = tmp; }
    const int last = (j == 5) ? 1 : 0;
    run_torgb(cur, H, Cout, 99 + 4 * j, skcur, skother, last);
    { float* tmp = skcur; skcur = skother; skother = tmp; }
  }
}
